// TransformerEncoder_6150393167969
// MI455X (gfx1250) — matrix operand order repaired and checked by reading, not compiled, not run on hardware
//
#include <hip/hip_runtime.h>

// ---------------------------------------------------------------------------
// Fused 2-layer transformer encoder for MI455X (gfx1250, wave32, WMMA + TDM).
//
// Roofline: ~1.65 TFLOP vs ~0.3 GB mandatory HBM traffic (~13us @ 23.3TB/s).
// bf16 WMMA makes compute negligible; the kernel is fully fused so activations
// never touch HBM. Weights are converted once per launch to fragment-swizzled
// bf16 in d_ws (12.6MB, L2-resident) so each B-operand is two contiguous
// global_load_b128 per lane. Each wave computes TWO N-tiles with independent
// accumulators to break the WMMA RAW chain and halve A-fragment LDS traffic.
// The per-WG x tile (64KB) is DMA'd into LDS by the Tensor Data Mover.
// ---------------------------------------------------------------------------

typedef unsigned short u16;
typedef unsigned int   u32;
typedef __attribute__((ext_vector_type(16))) __bf16 v16bf;
typedef __attribute__((ext_vector_type(8)))  float  v8f;
typedef __attribute__((ext_vector_type(4)))  u32    u32x4;
typedef __attribute__((ext_vector_type(8)))  int    i32x8;
typedef __attribute__((ext_vector_type(4)))  int    i32x4;

#if defined(__has_builtin)
#if __has_builtin(__builtin_amdgcn_tensor_load_to_lds) && \
    __has_builtin(__builtin_amdgcn_s_wait_tensorcnt)
#define HAS_TDM 1
#endif
#endif
#ifndef HAS_TDM
#define HAS_TDM 0
#endif

#define THREADS   256           // 8 wave32s
#define MROWS     4             // batch rows per workgroup
#define MTOK      32            // 4 rows * 8 tokens = 2 M-tiles of 16
#define DMODEL    512
#define DHID      2048
#define DK        64
#define HPITCH    520           // f32 h pitch (pad 8)
#define BPITCH    528           // bf16 pitch for 512-wide (pad 16)
#define HIDP      2080          // bf16 pitch for 2048-wide (pad 32)

// LDS layout (bytes); dynamic shared, 230KB < 320KB WGP budget
#define HF_BYTES  (MTOK*HPITCH*4)          // 66560  residual h (f32)
#define HB_BYTES  (MTOK*BPITCH*2)          // 33792  h (bf16) for A-fragments
#define SCR_OFF   (HF_BYTES + HB_BYTES)
#define SCR_BYTES 135168                   // max(QKV+ctx, FFN hidden, x stage)
#define SMEM_BYTES (SCR_OFF + SCR_BYTES)   // 235520

// ws layout (u16 elements): per layer Wq,Wk,Wv,Wo (512x512) then W1,W2
#define WMAT      262144                   // 512*512
#define WBIG      1048576                  // 512*2048
#define LSTRIDE   3145728                  // per-layer u16 count

__device__ __forceinline__ u16 f2bf(float f) {
  union { float f; u32 u; } v; v.f = f;
  u32 r = v.u + 0x7FFFu + ((v.u >> 16) & 1u);   // round-to-nearest-even
  return (u16)(r >> 16);
}
__device__ __forceinline__ float bf2f(u16 h) {
  union { u32 u; float f; } v; v.u = ((u32)h) << 16;
  return v.f;
}

// A-fragment (16x32 bf16) from LDS per ISA 7.12.2 "16-bit A-Matrix 16x32":
// lanes 0-15: VGPR j<4 -> K={2j,2j+1}, j>=4 -> K={16+2(j-4),...}; lanes 16-31: +8
__device__ __forceinline__ v16bf load_a_frag(const u16* A, int pitch,
                                             int mt, int kt, int lane) {
  const int m   = mt * 16 + (lane & 15);
  const int grp = lane >> 4;
  const u16* row = A + m * pitch + kt * 32;
  union { v16bf v; u32 u[8]; } U;
#pragma unroll
  for (int j = 0; j < 8; ++j) {
    int k = ((j < 4) ? (2 * j) : (16 + 2 * (j - 4))) + grp * 8;
    U.u[j] = *(const u32*)(row + k);        // 4B-aligned (k even, pitch even)
  }
  return U.v;
}

// B-fragment (32x16 bf16): pre-swizzled in global; lane reads 32 contiguous B.
__device__ __forceinline__ v16bf load_b_frag(const u16* Wf, int lane) {
  union { v16bf v; u32x4 q[2]; } U;
  const u32x4* p = (const u32x4*)Wf + lane * 2;
  U.q[0] = p[0];
  U.q[1] = p[1];
  return U.v;
}

// Two 16x16 output tiles (adjacent N-tiles nt0, nt0+1) sharing one A stream.
// Two independent accumulator chains hide WMMA XDL latency at low occupancy.
__device__ __forceinline__ void gemm_tile2(const u16* A, int apitch,
                                           const u16* W, int KT,
                                           int mt, int nt0, int lane,
                                           float bi0, float bi1,
                                           v8f& out0, v8f& out1) {
  v8f acc0 = { bi0, bi0, bi0, bi0, bi0, bi0, bi0, bi0 };
  v8f acc1 = { bi1, bi1, bi1, bi1, bi1, bi1, bi1, bi1 };
  const u16* wb0 = W + (size_t)nt0 * KT * 512;
  const u16* wb1 = wb0 + (size_t)KT * 512;
#pragma unroll 4
  for (int kt = 0; kt < KT; ++kt) {
    const u16* wf0 = wb0 + (size_t)kt * 512;
    const u16* wf1 = wb1 + (size_t)kt * 512;
    if ((kt & 3) == 0) {                    // global_prefetch 4KB ahead, WGP scope
      __builtin_prefetch(wf0 + 2048, 0, 3);
      __builtin_prefetch(wf1 + 2048, 0, 3);
    }
    v16bf a  = load_a_frag(A, apitch, mt, kt, lane);
    v16bf b0 = load_b_frag(wf0, lane);
    v16bf b1 = load_b_frag(wf1, lane);
    acc0 = __builtin_amdgcn_wmma_f32_16x16x32_bf16(
        false, a, false, b0, (short)0, acc0, false, false);
    acc1 = __builtin_amdgcn_wmma_f32_16x16x32_bf16(
        false, a, false, b1, (short)0, acc1, false, false);
  }
  out0 = acc0;
  out1 = acc1;
}

// f32 [K,N] row-major -> bf16 fragment-swizzled (ISA 16-bit B layout:
// lanes 0-15: VGPR j<4 -> K={2j,2j+1}, j>=4 -> K={16+2(j-4),...}; lanes 16-31: +8).
__global__ __launch_bounds__(256) void convert_swizzle(
    const float* __restrict__ src, u16* __restrict__ dst, int K, int N) {
  int i = blockIdx.x * 256 + threadIdx.x;       // one u32 (2 bf16) per thread
  int total = (K * N) >> 1;
  if (i >= total) return;
  int KT = K >> 5;
  int f = i >> 8;                                // fragment index = nt*KT + kt
  int q = i & 255;
  int lane = q >> 3, j = q & 7;
  int nt = f / KT, kt = f - nt * KT;
  int n = nt * 16 + (lane & 15);
  int k = kt * 32 + ((j < 4) ? (2 * j) : (16 + 2 * (j - 4))) + ((lane >> 4) << 3);
  u32 lo = f2bf(src[(size_t)k * N + n]);
  u32 hi = f2bf(src[(size_t)(k + 1) * N + n]);
  ((u32*)dst)[i] = lo | (hi << 16);
}

__device__ __forceinline__ void layernorm_inplace(float* hf, u16* hb,
                                                  const float* g, const float* b,
                                                  float* red, int tid) {
  const int m = tid >> 3, p = tid & 7, d0 = p * 64;
  float* row = hf + m * HPITCH;
  float s = 0.f, ss = 0.f;
  for (int d = d0; d < d0 + 64; ++d) { float v = row[d]; s += v; ss += v * v; }
  red[(m * 8 + p) * 2 + 0] = s;
  red[(m * 8 + p) * 2 + 1] = ss;
  __syncthreads();
  float mu = 0.f, sq = 0.f;
  for (int q = 0; q < 8; ++q) { mu += red[(m * 8 + q) * 2]; sq += red[(m * 8 + q) * 2 + 1]; }
  mu *= (1.f / 512.f);
  float var = sq * (1.f / 512.f) - mu * mu;
  float rs = rsqrtf(var + 1e-6f);
  u16* brow = hb + m * BPITCH;
  for (int d = d0; d < d0 + 64; ++d) {
    float v = (row[d] - mu) * rs * g[d] + b[d];
    row[d] = v;
    brow[d] = f2bf(v);
  }
  __syncthreads();
}

__global__ __launch_bounds__(THREADS) void encoder_kernel(
    const float* __restrict__ x, const u16* __restrict__ wsb,
    const float* __restrict__ bq, const float* __restrict__ bk,
    const float* __restrict__ bv, const float* __restrict__ bo,
    const float* __restrict__ g1, const float* __restrict__ be1,
    const float* __restrict__ b1, const float* __restrict__ b2,
    const float* __restrict__ g2, const float* __restrict__ be2,
    float* __restrict__ out) {
  extern __shared__ char smem[];
  float* hf = (float*)smem;                      // [32][520] residual (f32)
  u16*   hb = (u16*)(smem + HF_BYTES);           // [32][528] h (bf16)
  u16*   scr = (u16*)(smem + SCR_OFF);

  const int tid = threadIdx.x;
  const int lane = tid & 31;
  const int wave = tid >> 5;
  const int blk = blockIdx.x;

#if HAS_TDM
  // ---- TDM: DMA the 32x512 f32 x-tile into the LDS scratch region.
  // D# group0: count=1 | lds_addr | global_addr(57b) | type=2
  // D# group1: data_size=4B, tensor 512x32 (stride 512), tile 512x32
  if (wave == 0) {
    unsigned long long ga =
        (unsigned long long)(const void*)(x + (size_t)blk * MTOK * DMODEL);
    u32x4 g0;
    g0[0] = 1u;                                    // count=1 (valid user D#)
    g0[1] = (u32)SCR_OFF;                          // lds_addr (bytes)
    g0[2] = (u32)(ga & 0xFFFFFFFFull);             // global_addr[31:0]
    g0[3] = (u32)((ga >> 32) & 0x1FFFFFFull) | (2u << 30);  // addr[56:32]|type=2
    i32x8 g1v;
    g1v[0] = 0x20000;                              // data_size=2 (4 bytes)
    g1v[1] = (int)(512u << 16);                    // tensor_dim0 = 512
    g1v[2] = (int)(32u << 16);                     // tensor_dim1 = 32
    g1v[3] = (int)(512u << 16);                    // tile_dim0 = 512
    g1v[4] = 32;                                   // tile_dim1 = 32
    g1v[5] = 512;                                  // tensor_dim0_stride = 512
    g1v[6] = 0;
    g1v[7] = 0;
    i32x4 z4 = { 0, 0, 0, 0 };                     // groups 2/3 unused (2-D)
    i32x8 z8 = { 0, 0, 0, 0, 0, 0, 0, 0 };         // trailing group (6-arg form)
    __builtin_amdgcn_tensor_load_to_lds(g0, g1v, z4, z4, z8, 0);
    __builtin_amdgcn_s_wait_tensorcnt(0);
  }
  __syncthreads();
#endif

  // ---- stage 0: x + sinusoidal positional encoding -> hf (f32) + hb (bf16)
  for (int i = tid; i < MTOK * DMODEL; i += THREADS) {
    int m = i >> 9, d = i & 511;
    int s = m & 7;
    float fd = (float)(2 * (d >> 1));
    float ang = (float)s * __powf(10000.f, -fd * (1.f / 512.f));
    float pe = (d & 1) ? __cosf(ang) : __sinf(ang);
#if HAS_TDM
    float v = ((const float*)scr)[m * DMODEL + d] + pe;
#else
    float v = x[(size_t)(blk * MTOK + m) * DMODEL + d] + pe;
#endif
    hf[m * HPITCH + d] = v;
    hb[m * BPITCH + d] = f2bf(v);
  }
  __syncthreads();

  for (int l = 0; l < 2; ++l) {
    const u16* wq = wsb + (size_t)l * LSTRIDE;
    const u16* wk = wq + WMAT;
    const u16* wv = wk + WMAT;
    const u16* wo = wv + WMAT;
    const u16* w1 = wo + WMAT;
    const u16* w2 = w1 + WBIG;
    const float* bq_l = bq + l * DMODEL;
    const float* bk_l = bk + l * DMODEL;
    const float* bv_l = bv + l * DMODEL;
    const float* bo_l = bo + l * DMODEL;
    const float* g1_l = g1 + l * DMODEL;
    const float* e1_l = be1 + l * DMODEL;
    const float* b1_l = b1 + l * DHID;
    const float* b2_l = b2 + l * DMODEL;
    const float* g2_l = g2 + l * DMODEL;
    const float* e2_l = be2 + l * DMODEL;

    u16* qkv = scr;                              // [3][32][528] bf16
    u16* ctx = scr + 3 * MTOK * BPITCH;          // [32][528] bf16

    // ---- QKV projections: 96 tile-pairs across 8 waves
    for (int t = wave; t < 96; t += 8) {
      int mt = t & 1, np = t >> 1;               // np in 0..47
      int nt0 = np * 2;                          // pairs never straddle a matrix
      int mat = nt0 >> 5, nc0 = nt0 & 31;
      const u16* W = (mat == 0) ? wq : (mat == 1) ? wk : wv;
      const float* bias = (mat == 0) ? bq_l : (mat == 1) ? bk_l : bv_l;
      float bi0 = bias[nc0 * 16 + (lane & 15)];
      float bi1 = bias[(nc0 + 1) * 16 + (lane & 15)];
      v8f a0, a1;
      gemm_tile2(hb, BPITCH, W, 16, mt, nc0, lane, bi0, bi1, a0, a1);
      u16* dst = qkv + mat * MTOK * BPITCH;
      int n0 = nc0 * 16 + (lane & 15);
      int mbase = mt * 16 + ((lane >> 4) << 3);
#pragma unroll
      for (int r = 0; r < 8; ++r) {
        dst[(mbase + r) * BPITCH + n0]      = f2bf(a0[r]);
        dst[(mbase + r) * BPITCH + n0 + 16] = f2bf(a1[r]);
      }
    }
    __syncthreads();

    // ---- attention: seq=8 within each batch row; one thread per (token,head)
    {
      int m = tid >> 3, hd = tid & 7;
      const u16* Q = qkv;
      const u16* Kb = qkv + MTOK * BPITCH;
      const u16* Vb = qkv + 2 * MTOK * BPITCH;
      int rbase = m & ~7;
      const u16* qrow = Q + m * BPITCH + hd * DK;
      float p[8];
      float mx = -1e30f;
      for (int j = 0; j < 8; ++j) {
        const u16* krow = Kb + (rbase + j) * BPITCH + hd * DK;
        float s = 0.f;
        for (int d = 0; d < DK; ++d) s += bf2f(qrow[d]) * bf2f(krow[d]);
        s *= 0.125f;                             // 1/sqrt(64)
        p[j] = s;
        mx = fmaxf(mx, s);
      }
      float den = 0.f;
      for (int j = 0; j < 8; ++j) { p[j] = __expf(p[j] - mx); den += p[j]; }
      float inv = 1.f / den;
      for (int j = 0; j < 8; ++j) p[j] *= inv;
      u16* crow = ctx + m * BPITCH + hd * DK;
      for (int d = 0; d < DK; ++d) {
        float s = 0.f;
        for (int j = 0; j < 8; ++j)
          s += p[j] * bf2f(Vb[(rbase + j) * BPITCH + hd * DK + d]);
        crow[d] = f2bf(s);
      }
    }
    __syncthreads();

    // ---- output projection + residual add (pre-LN result into hf)
    for (int t = wave; t < 32; t += 8) {
      int mt = t & 1, np = t >> 1;
      int nt0 = np * 2;
      float bi0 = bo_l[nt0 * 16 + (lane & 15)];
      float bi1 = bo_l[(nt0 + 1) * 16 + (lane & 15)];
      v8f a0, a1;
      gemm_tile2(ctx, BPITCH, wo, 16, mt, nt0, lane, bi0, bi1, a0, a1);
      int n0 = nt0 * 16 + (lane & 15);
      int mbase = mt * 16 + ((lane >> 4) << 3);
#pragma unroll
      for (int r = 0; r < 8; ++r) {
        hf[(mbase + r) * HPITCH + n0]      += a0[r];
        hf[(mbase + r) * HPITCH + n0 + 16] += a1[r];
      }
    }
    __syncthreads();
    layernorm_inplace(hf, hb, g1_l, e1_l, (float*)scr, tid);

    // ---- FFN1: [32,512]@[512,2048] + bias, ReLU -> bf16 hidden in LDS
    u16* hid = scr;                              // [32][2080] bf16
    for (int t = wave; t < 128; t += 8) {
      int mt = t & 1, np = t >> 1;               // np in 0..63
      int nt0 = np * 2;
      float bi0 = b1_l[nt0 * 16 + (lane & 15)];
      float bi1 = b1_l[(nt0 + 1) * 16 + (lane & 15)];
      v8f a0, a1;
      gemm_tile2(hb, BPITCH, w1, 16, mt, nt0, lane, bi0, bi1, a0, a1);
      int n0 = nt0 * 16 + (lane & 15);
      int mbase = mt * 16 + ((lane >> 4) << 3);
#pragma unroll
      for (int r = 0; r < 8; ++r) {
        hid[(mbase + r) * HIDP + n0]      = f2bf(fmaxf(a0[r], 0.f));
        hid[(mbase + r) * HIDP + n0 + 16] = f2bf(fmaxf(a1[r], 0.f));
      }
    }
    __syncthreads();

    // ---- FFN2: [32,2048]@[2048,512] + bias + residual
    for (int t = wave; t < 32; t += 8) {
      int mt = t & 1, np = t >> 1;
      int nt0 = np * 2;
      float bi0 = b2_l[nt0 * 16 + (lane & 15)];
      float bi1 = b2_l[(nt0 + 1) * 16 + (lane & 15)];
      v8f a0, a1;
      gemm_tile2(hid, HIDP, w2, 64, mt, nt0, lane, bi0, bi1, a0, a1);
      int n0 = nt0 * 16 + (lane & 15);
      int mbase = mt * 16 + ((lane >> 4) << 3);
#pragma unroll
      for (int r = 0; r < 8; ++r) {
        hf[(mbase + r) * HPITCH + n0]      += a0[r];
        hf[(mbase + r) * HPITCH + n0 + 16] += a1[r];
      }
    }
    __syncthreads();
    layernorm_inplace(hf, hb, g2_l, e2_l, (float*)scr, tid);
  }

  // ---- output gather: out[b] = h[b, b % 8]
  for (int i = tid; i < MROWS * DMODEL; i += THREADS) {
    int rl = i >> 9, d = i & 511;
    int br = blk * MROWS + rl;
    int m = rl * 8 + (br & 7);
    out[(size_t)br * DMODEL + d] = hf[m * HPITCH + d];
  }
}

extern "C" void kernel_launch(void* const* d_in, const int* in_sizes, int n_in,
                              void* d_out, int out_size, void* d_ws, size_t ws_size,
                              hipStream_t stream) {
  (void)in_sizes; (void)n_in; (void)out_size; (void)ws_size;
  const float* x  = (const float*)d_in[0];
  const float* Wq = (const float*)d_in[1];
  const float* bq = (const float*)d_in[2];
  const float* Wk = (const float*)d_in[3];
  const float* bk = (const float*)d_in[4];
  const float* Wv = (const float*)d_in[5];
  const float* bv = (const float*)d_in[6];
  const float* Wo = (const float*)d_in[7];
  const float* bo = (const float*)d_in[8];
  const float* g1 = (const float*)d_in[9];
  const float* e1 = (const float*)d_in[10];
  const float* W1 = (const float*)d_in[11];
  const float* b1 = (const float*)d_in[12];
  const float* W2 = (const float*)d_in[13];
  const float* b2 = (const float*)d_in[14];
  const float* g2 = (const float*)d_in[15];
  const float* e2 = (const float*)d_in[16];
  float* out = (float*)d_out;
  u16* wsb = (u16*)d_ws;                  // needs 12.6 MB of scratch

  // Per-launch weight convert+swizzle (deterministic; inputs untouched).
  struct Job { const float* src; size_t dst; int K, N; };
  Job jobs[12];
  int nj = 0;
  for (int l = 0; l < 2; ++l) {
    size_t base = (size_t)l * LSTRIDE;
    jobs[nj++] = { Wq + (size_t)l * WMAT, base + 0 * WMAT, 512, 512 };
    jobs[nj++] = { Wk + (size_t)l * WMAT, base + 1 * WMAT, 512, 512 };
    jobs[nj++] = { Wv + (size_t)l * WMAT, base + 2 * WMAT, 512, 512 };
    jobs[nj++] = { Wo + (size_t)l * WMAT, base + 3 * WMAT, 512, 512 };
    jobs[nj++] = { W1 + (size_t)l * WBIG, base + 4 * WMAT, 512, 2048 };
    jobs[nj++] = { W2 + (size_t)l * WBIG, base + 4 * WMAT + WBIG, 2048, 512 };
  }
  for (int j = 0; j < nj; ++j) {
    int pairs = (jobs[j].K * jobs[j].N) >> 1;
    int blocks = (pairs + 255) / 256;
    convert_swizzle<<<blocks, 256, 0, stream>>>(jobs[j].src, wsb + jobs[j].dst,
                                                jobs[j].K, jobs[j].N);
  }

  (void)hipFuncSetAttribute((const void*)encoder_kernel,
                            hipFuncAttributeMaxDynamicSharedMemorySize,
                            SMEM_BYTES);
  encoder_kernel<<<16384 / MROWS, THREADS, SMEM_BYTES, stream>>>(
      x, wsb, bq, bk, bv, bo, g1, e1, b1, b2, g2, e2, out);
}
